// DGT_44822278701844
// MI455X (gfx1250) — matrix operand order repaired and checked by reading, not compiled, not run on hardware
//
#include <hip/hip_runtime.h>
#include <hip/hip_bf16.h>
#include <math.h>

#define NNODES 50000
#define NEDGES 300000
#define HID    256
#define HEADS  8
#define CH     32

typedef __attribute__((ext_vector_type(16))) __bf16 v16bf;
typedef __attribute__((ext_vector_type(8)))  float  v8f;

// ---------- helpers ----------
__device__ __forceinline__ unsigned short f2bf(float x) {
    unsigned u = __float_as_uint(x);
    u += 0x7FFFu + ((u >> 16) & 1u);           // round to nearest even
    return (unsigned short)(u >> 16);
}
// monotone order-preserving float<->uint encoding for atomicMax
__device__ __forceinline__ unsigned fenc(float f) {
    unsigned u = __float_as_uint(f);
    return (u & 0x80000000u) ? ~u : (u | 0x80000000u);
}
__device__ __forceinline__ float fdec(unsigned u) {
    unsigned v = (u & 0x80000000u) ? (u & 0x7FFFFFFFu) : ~u;
    return __uint_as_float(v);
}

// ---------- f32 -> bf16 cast ----------
__global__ void k_cast_bf16(const float* __restrict__ in,
                            unsigned short* __restrict__ out, int n) {
    int i = blockIdx.x * blockDim.x + threadIdx.x;
    if (i < n) out[i] = f2bf(in[i]);
}

// ---------- pack W [256x256] f32 row-major (k-major) into WMMA B-fragment layout ----------
// packed index = ((nt*8 + kt)*32 + lane)*16 + j
//   k = kt*32 + (lane>>4)*16 + j ; n = nt*16 + (lane&15)
__global__ void k_pack_w(const float* __restrict__ W,
                         unsigned short* __restrict__ Bp) {
    int tid = blockIdx.x * blockDim.x + threadIdx.x;   // 0 .. 65535
    int j    = tid & 15;
    int lane = (tid >> 4) & 31;
    int kt   = (tid >> 9) & 7;
    int nt   = tid >> 12;
    int k = kt * 32 + ((lane >> 4) << 4) + j;
    int n = nt * 16 + (lane & 15);
    Bp[tid] = f2bf(W[k * HID + n]);
}

// ---------- WMMA GEMM: Y[M,256] = A(bf16)[M,256] @ Wpacked + bias ----------
__global__ __launch_bounds__(128)
void k_gemm_bias(const unsigned short* __restrict__ A,
                 const unsigned short* __restrict__ Bp,
                 const float* __restrict__ bias,
                 float* __restrict__ Y) {
    const int lane = threadIdx.x & 31;
    const int wave = threadIdx.x >> 5;        // 0..3, owns cols [wave*64, wave*64+64)
    const int m0   = blockIdx.x << 4;
    const int half = lane >> 4;               // lane group (K interleave select)
    const int mrow = lane & 15;

    const unsigned short* arow = A + (size_t)(m0 + mrow) * HID;

    v8f acc[4] = {};

    for (int kt = 0; kt < 8; ++kt) {
        // A fragment: 16x32 bf16, contiguous K order (as packed B) — lane holds row mrow,
        // elements j=0..15 -> K = kt*32 + half*16 + j
        union { uint4 q[2]; v16bf v; } af;
        const uint4* ap = reinterpret_cast<const uint4*>(arow + kt * 32 + half * 16);
        af.q[0] = ap[0];   // 8 bf16 at K = kt*32 + half*16
        af.q[1] = ap[1];   // 8 bf16 at K = kt*32 + half*16 + 8
        #pragma unroll
        for (int t = 0; t < 4; ++t) {
            int nt = wave * 4 + t;
            union { uint4 q[2]; v16bf v; } bfr;
            const uint4* bp = reinterpret_cast<const uint4*>(
                Bp + ((size_t)(nt * 8 + kt) * 32 + lane) * 16);
            bfr.q[0] = bp[0];
            bfr.q[1] = bp[1];
            acc[t] = __builtin_amdgcn_wmma_f32_16x16x32_bf16(
                false, af.v, false, bfr.v, (short)0, acc[t], false, false);
        }
    }

    #pragma unroll
    for (int t = 0; t < 4; ++t) {
        int col = wave * 64 + t * 16 + (lane & 15);
        float b = bias[col];
        #pragma unroll
        for (int r = 0; r < 8; ++r) {
            int row = m0 + half * 8 + r;     // C/D layout: VGPR r -> M = r + half*8
            Y[(size_t)row * HID + col] = acc[t][r] + b;
        }
    }
}

// ---------- per-layer edge-feature collapse: u = timeW @ We ; c = timeB @ We + be ----------
__global__ void k_uc(const float* __restrict__ tw, const float* __restrict__ tb,
                     const float* __restrict__ We, const float* __restrict__ be,
                     float* __restrict__ u, float* __restrict__ c) {
    int t = threadIdx.x;                      // 0..255
    float su = 0.f, sc = 0.f;
    for (int i = 0; i < HID; ++i) {
        float w = We[i * HID + t];
        su = fmaf(tw[i], w, su);
        sc = fmaf(tb[i], w, sc);
    }
    u[t] = su;
    c[t] = sc + be[t];
}

// ---------- init segment buffers ----------
__global__ void k_init_seg(unsigned* __restrict__ amaxu, float* __restrict__ denom, int n) {
    int i = blockIdx.x * blockDim.x + threadIdx.x;
    if (i < n) { amaxu[i] = 0u; denom[i] = 0.f; }
}

// ---------- alpha = q[dst] . (k[src] + t*u + c) / sqrt(C), segment max ----------
__global__ void k_alpha(const float* __restrict__ q, const float* __restrict__ k,
                        const int* __restrict__ src, const int* __restrict__ dst,
                        const float* __restrict__ et,
                        const float* __restrict__ u, const float* __restrict__ c,
                        float* __restrict__ alpha, unsigned* __restrict__ amaxu) {
    int idx = blockIdx.x * blockDim.x + threadIdx.x;
    if (idx >= NEDGES * HEADS) return;
    int e = idx >> 3, h = idx & 7;
    int s = src[e], d = dst[e];
    float t = et[e];
    const float* qr = q + (size_t)d * HID + h * CH;
    const float* kr = k + (size_t)s * HID + h * CH;
    const float* ur = u + h * CH;
    const float* cr = c + h * CH;
    float acc = 0.f;
    #pragma unroll
    for (int i = 0; i < CH; ++i)
        acc = fmaf(qr[i], kr[i] + fmaf(t, ur[i], cr[i]), acc);
    acc *= 0.17677669529663687f;              // 1/sqrt(32)
    alpha[idx] = acc;
    atomicMax(&amaxu[(size_t)d * HEADS + h], fenc(acc));
}

// ---------- w = exp(alpha - amax[dst]); segment sum ----------
__global__ void k_expw(const int* __restrict__ dst,
                       float* __restrict__ alpha,
                       const unsigned* __restrict__ amaxu,
                       float* __restrict__ denom) {
    int idx = blockIdx.x * blockDim.x + threadIdx.x;
    if (idx >= NEDGES * HEADS) return;
    int e = idx >> 3, h = idx & 7;
    int d = dst[e];
    float m = fdec(amaxu[(size_t)d * HEADS + h]);
    float w = __expf(alpha[idx] - m);
    alpha[idx] = w;
    atomicAdd(&denom[(size_t)d * HEADS + h], w);
}

// ---------- msg = (v[src] + t*u + c) * attn, scatter-add into accum (pre-loaded with skip) ----------
__global__ void k_msg(const float* __restrict__ v,
                      const int* __restrict__ src, const int* __restrict__ dst,
                      const float* __restrict__ et,
                      const float* __restrict__ u, const float* __restrict__ c,
                      const float* __restrict__ alpha, const float* __restrict__ denom,
                      float* __restrict__ accum) {
    int idx = blockIdx.x * blockDim.x + threadIdx.x;
    if (idx >= NEDGES * HEADS) return;
    int e = idx >> 3, h = idx & 7;
    int s = src[e], d = dst[e];
    float attn = alpha[idx] / denom[(size_t)d * HEADS + h];
    float t = et[e];
    const float* vr = v + (size_t)s * HID + h * CH;
    const float* ur = u + h * CH;
    const float* cr = c + h * CH;
    float* o = accum + (size_t)d * HID + h * CH;
    #pragma unroll
    for (int i = 0; i < CH; ++i)
        atomicAdd(&o[i], (vr[i] + fmaf(t, ur[i], cr[i])) * attn);
}

// ---------- relu + f32 copy + bf16 copy for next layer's WMMA A ----------
__global__ void k_finish(const float* __restrict__ accum,
                         float* __restrict__ hcur,
                         unsigned short* __restrict__ hbf, int n) {
    int i = blockIdx.x * blockDim.x + threadIdx.x;
    if (i >= n) return;
    float r = accum[i];
    r = r > 0.f ? r : 0.f;
    hcur[i] = r;
    hbf[i] = f2bf(r);
}

// ---------- final classifier: [N,256] x [256,2] ----------
__global__ void k_out(const float* __restrict__ h,
                      const float* __restrict__ W, const float* __restrict__ b,
                      float* __restrict__ out) {
    int n = blockIdx.x * blockDim.x + threadIdx.x;
    if (n >= NNODES) return;
    const float* hr = h + (size_t)n * HID;
    float a0 = b[0], a1 = b[1];
    for (int i = 0; i < HID; ++i) {
        float x = hr[i];
        a0 = fmaf(x, W[i * 2 + 0], a0);
        a1 = fmaf(x, W[i * 2 + 1], a1);
    }
    out[n * 2 + 0] = a0;
    out[n * 2 + 1] = a1;
}

extern "C" void kernel_launch(void* const* d_in, const int* in_sizes, int n_in,
                              void* d_out, int out_size, void* d_ws, size_t ws_size,
                              hipStream_t stream) {
    const float* x     = (const float*)d_in[0];
    const int*   ei    = (const int*)d_in[1];     // [2,E] (src row, dst row)
    const float* et    = (const float*)d_in[2];
    const float* timeW = (const float*)d_in[3];   // [1,256] -> row vector
    const float* timeB = (const float*)d_in[4];
    const float* outW  = (const float*)d_in[5];   // [256,2]
    const float* outB  = (const float*)d_in[6];
    const int* src = ei;
    const int* dst = ei + NEDGES;

    // workspace carve-up
    char* ws = (char*)d_ws;
    size_t off = 0;
    auto alloc = [&](size_t bytes) -> char* {
        char* p = ws + off;
        off += (bytes + 255) & ~(size_t)255;
        return p;
    };
    unsigned short* hbf   = (unsigned short*)alloc((size_t)NNODES * HID * 2);
    float*          qb    = (float*)alloc((size_t)NNODES * HID * 4);
    float*          kb    = (float*)alloc((size_t)NNODES * HID * 4);
    float*          vb    = (float*)alloc((size_t)NNODES * HID * 4);
    float*          accum = (float*)alloc((size_t)NNODES * HID * 4);
    float*          hcur  = (float*)alloc((size_t)NNODES * HID * 4);
    float*          alpha = (float*)alloc((size_t)NEDGES * HEADS * 4);
    unsigned*       amaxu = (unsigned*)alloc((size_t)NNODES * HEADS * 4);
    float*          denom = (float*)alloc((size_t)NNODES * HEADS * 4);
    unsigned short* packW = (unsigned short*)alloc((size_t)HID * HID * 2);
    float*          uvec  = (float*)alloc(HID * 4);
    float*          cvec  = (float*)alloc(HID * 4);

    const int EH_BLOCKS = (NEDGES * HEADS) / 256;          // 9375 exact
    const int NH_BLOCKS = (NNODES * HEADS + 255) / 256;
    const int NE_BLOCKS = (NNODES * HID) / 256;            // 50000 exact
    const int PK_BLOCKS = (HID * HID) / 256;               // 256

    // h0 = x (bf16 for WMMA A)
    k_cast_bf16<<<NE_BLOCKS, 256, 0, stream>>>(x, hbf, NNODES * HID);

    for (int l = 0; l < 2; ++l) {
        const float* Wq = (const float*)d_in[7 + l * 10 + 0];
        const float* bq = (const float*)d_in[7 + l * 10 + 1];
        const float* Wk = (const float*)d_in[7 + l * 10 + 2];
        const float* bk = (const float*)d_in[7 + l * 10 + 3];
        const float* Wv = (const float*)d_in[7 + l * 10 + 4];
        const float* bv = (const float*)d_in[7 + l * 10 + 5];
        const float* We = (const float*)d_in[7 + l * 10 + 6];
        const float* be = (const float*)d_in[7 + l * 10 + 7];
        const float* Ws = (const float*)d_in[7 + l * 10 + 8];
        const float* bs = (const float*)d_in[7 + l * 10 + 9];

        // q/k/v projections + skip (skip GEMM seeds the aggregation accumulator)
        k_pack_w<<<PK_BLOCKS, 256, 0, stream>>>(Wq, packW);
        k_gemm_bias<<<NNODES / 16, 128, 0, stream>>>(hbf, packW, bq, qb);
        k_pack_w<<<PK_BLOCKS, 256, 0, stream>>>(Wk, packW);
        k_gemm_bias<<<NNODES / 16, 128, 0, stream>>>(hbf, packW, bk, kb);
        k_pack_w<<<PK_BLOCKS, 256, 0, stream>>>(Wv, packW);
        k_gemm_bias<<<NNODES / 16, 128, 0, stream>>>(hbf, packW, bv, vb);
        k_pack_w<<<PK_BLOCKS, 256, 0, stream>>>(Ws, packW);
        k_gemm_bias<<<NNODES / 16, 128, 0, stream>>>(hbf, packW, bs, accum);

        // rank-1 edge-feature collapse: e = t*u + c
        k_uc<<<1, 256, 0, stream>>>(timeW, timeB, We, be, uvec, cvec);

        // segment softmax + aggregation
        k_init_seg<<<NH_BLOCKS, 256, 0, stream>>>(amaxu, denom, NNODES * HEADS);
        k_alpha<<<EH_BLOCKS, 256, 0, stream>>>(qb, kb, src, dst, et, uvec, cvec, alpha, amaxu);
        k_expw<<<EH_BLOCKS, 256, 0, stream>>>(dst, alpha, amaxu, denom);
        k_msg<<<EH_BLOCKS, 256, 0, stream>>>(vb, src, dst, et, uvec, cvec, alpha, denom, accum);

        // relu(accum) -> hcur (f32) + hbf (bf16 for next layer)
        k_finish<<<NE_BLOCKS, 256, 0, stream>>>(accum, hcur, hbf, NNODES * HID);
    }

    k_out<<<(NNODES + 255) / 256, 256, 0, stream>>>(hcur, outW, outB, (float*)d_out);
}
